// CausalSelfAttention_73263552135318
// MI455X (gfx1250) — matrix operand order repaired and checked by reading, not compiled, not run on hardware
//
#include <hip/hip_runtime.h>

// Problem constants (match reference)
#define BB 4
#define TT 2048
#define DD 1024
#define HH 16
#define DHc 64
#define MM (BB * TT)     // 8192
#define NQKV (3 * DD)    // 3072

typedef __attribute__((ext_vector_type(16))) __bf16 v16bf;
typedef __attribute__((ext_vector_type(8)))  float  v8f;

union FragBF {
    v16bf v;
    uint4 q[2];
    unsigned short s[16];
};

__device__ __forceinline__ unsigned short f2bf(float f) {
    unsigned int u = __builtin_bit_cast(unsigned int, f);
    u += 0x7FFFu + ((u >> 16) & 1u);   // round-to-nearest-even
    return (unsigned short)(u >> 16);
}

__device__ __forceinline__ v8f vzero8() {
    v8f z = {0.f, 0.f, 0.f, 0.f, 0.f, 0.f, 0.f, 0.f};
    return z;
}

// ---------------------------------------------------------------------------
// Kernel 0: fp32 -> bf16 conversion (vectorized 4-wide)
// ---------------------------------------------------------------------------
__global__ void cvt_f32_to_bf16(const float* __restrict__ src,
                                unsigned short* __restrict__ dst, int n4) {
    int i = blockIdx.x * blockDim.x + threadIdx.x;
    int stride = gridDim.x * blockDim.x;
    for (; i < n4; i += stride) {
        float4 f = ((const float4*)src)[i];
        ushort4 o;
        o.x = f2bf(f.x); o.y = f2bf(f.y); o.z = f2bf(f.z); o.w = f2bf(f.w);
        ((ushort4*)dst)[i] = o;
    }
}

// ---------------------------------------------------------------------------
// Kernels 1 & 3: bf16 GEMM  C(M,N) = A(M,K) * W(N,K)^T + bias
// Block tile 64x64, 4 waves (128 threads); wave w owns rows [w*16, w*16+16).
// MODE 0: scatter qkv output (bf16) into per-head Q/K/V, scale Q by 1/sqrt(dh)
// MODE 1: write fp32 C + bias to outF (final projection)
// ---------------------------------------------------------------------------
template <int MODE>
__global__ __launch_bounds__(128) void gemm_bf16_wmma(
    const unsigned short* __restrict__ A, const unsigned short* __restrict__ W,
    const float* __restrict__ bias, int N, int K, float* __restrict__ outF,
    unsigned short* __restrict__ Qb, unsigned short* __restrict__ Kb,
    unsigned short* __restrict__ Vb) {
    __shared__ unsigned short As[64][40];   // 64 rows x 32 halves (+8 pad)
    __shared__ unsigned short Bs[64][40];

    const int m0 = blockIdx.x * 64;
    const int n0 = blockIdx.y * 64;
    const int tid = threadIdx.x;
    const int w = tid >> 5;
    const int lane = tid & 31;
    const int col = lane & 15;
    const int hf = lane >> 4;

    v8f acc[4];
#pragma unroll
    for (int nt = 0; nt < 4; ++nt) acc[nt] = vzero8();

    const int srow = tid >> 1;            // 0..63
    const int sc0 = (tid & 1) * 16;       // 0 or 16 halves
    const unsigned short* gA = A + (size_t)(m0 + srow) * K + sc0;
    const unsigned short* gW = W + (size_t)(n0 + srow) * K + sc0;

    for (int k0 = 0; k0 < K; k0 += 32) {
        __syncthreads();
        {
            const uint4* pa = (const uint4*)(gA + k0);
            *(uint4*)&As[srow][sc0]     = pa[0];
            *(uint4*)&As[srow][sc0 + 8] = pa[1];
            const uint4* pw = (const uint4*)(gW + k0);
            *(uint4*)&Bs[srow][sc0]     = pw[0];
            *(uint4*)&Bs[srow][sc0 + 8] = pw[1];
            if (k0 + 32 < K) {
                __builtin_prefetch(gA + k0 + 32, 0, 1);
                __builtin_prefetch(gW + k0 + 32, 0, 1);
            }
        }
        __syncthreads();

        // A fragment: 16x32 (rows w*16..), contiguous k = 16*hf + s per lane
        FragBF a;
        const int ar = (w << 4) + col;
        const int ac = hf << 4;           // 0 or 16
        a.q[0] = *(const uint4*)&As[ar][ac];
        a.q[1] = *(const uint4*)&As[ar][ac + 8];

#pragma unroll
        for (int nt = 0; nt < 4; ++nt) {
            // B fragment: 32x16, element(k,n) = W[n][k]; contiguous per lane
            FragBF b;
            const int br = (nt << 4) + col;
            const int bc = hf << 4;       // 0 or 16
            b.q[0] = *(const uint4*)&Bs[br][bc];
            b.q[1] = *(const uint4*)&Bs[br][bc + 8];
            acc[nt] = __builtin_amdgcn_wmma_f32_16x16x32_bf16(
                false, a.v, false, b.v, (short)0, acc[nt], false, false);
        }
    }

    // Epilogue. C layout: VGPR r, lanes 0-15 -> M=r, lanes 16-31 -> M=r+8; N=col.
    if (MODE == 1) {
#pragma unroll
        for (int nt = 0; nt < 4; ++nt) {
            const int n = n0 + nt * 16 + col;
            const float bv = bias[n];
#pragma unroll
            for (int r = 0; r < 8; ++r) {
                const int mi = m0 + (w << 4) + r + hf * 8;
                outF[(size_t)mi * N + n] = acc[nt][r] + bv;
            }
        }
    } else {
#pragma unroll
        for (int nt = 0; nt < 4; ++nt) {
            const int n = n0 + nt * 16 + col;
            const float bv = bias[n];
            const int which = n >> 10;            // 0=q 1=k 2=v
            const int rem = n & 1023;
            const int head = rem >> 6;
            const int dh = rem & 63;
            unsigned short* dst = (which == 0) ? Qb : ((which == 1) ? Kb : Vb);
            const float scl = (which == 0) ? 0.125f : 1.0f;  // 1/sqrt(64)
#pragma unroll
            for (int r = 0; r < 8; ++r) {
                const int mi = m0 + (w << 4) + r + hf * 8;
                const int b = mi >> 11;           // / TT
                const int t = mi & 2047;
                const size_t idx =
                    (((size_t)(b * HH + head)) * TT + t) * DHc + dh;
                dst[idx] = f2bf((acc[nt][r] + bv) * scl);
            }
        }
    }
}

// ---------------------------------------------------------------------------
// Kernel 2: flash attention. One workgroup = 64 query rows of one (b,h).
// 4 waves, each owns 16 query rows. Online softmax, bf16 WMMA, f32 accum.
// ---------------------------------------------------------------------------
__global__ __launch_bounds__(128) void flash_attn_wmma(
    const unsigned short* __restrict__ Qb, const unsigned short* __restrict__ Kb,
    const unsigned short* __restrict__ Vb, const unsigned char* __restrict__ amask,
    unsigned short* __restrict__ Yb) {
    __shared__ unsigned short Ks[64][72];       // [key][dh]   (+8 pad)
    __shared__ unsigned short Vt[64][72];       // [dh][key]   transposed
    __shared__ unsigned short Ps[4][16][72];    // per-wave P round-trip

    const int bh = blockIdx.y;                  // b*16 + h
    const int b = bh >> 4;
    const int h = bh & 15;
    const int q0 = blockIdx.x * 64;
    const int tid = threadIdx.x;
    const int w = tid >> 5;
    const int lane = tid & 31;
    const int col = lane & 15;
    const int hf = lane >> 4;

    // Q fragments for this wave's 16 rows: 2 x (16x32) A-fragments, kept live.
    const unsigned short* Qbase =
        Qb + ((size_t)bh * TT + q0 + (w << 4) + col) * DHc;
    FragBF qf[2];
#pragma unroll
    for (int f = 0; f < 2; ++f) {
        const int c = f * 32 + hf * 16;
        qf[f].q[0] = *(const uint4*)(Qbase + c);
        qf[f].q[1] = *(const uint4*)(Qbase + c + 8);
    }

    v8f Ot[4];
    float mrow[8], lrow[8];
#pragma unroll
    for (int nt = 0; nt < 4; ++nt) Ot[nt] = vzero8();
#pragma unroll
    for (int r = 0; r < 8; ++r) { mrow[r] = -1e30f; lrow[r] = 0.f; }

    const int kbmax = blockIdx.x;               // causal: last key block
    for (int kb = 0; kb <= kbmax; ++kb) {
        __syncthreads();
        // Stage K (row-major) and V (transposed) into LDS: 128 threads,
        // each handles half a 64-half row (32 halves = 4x b128 loads).
        {
            const int row = tid >> 1;
            const int c0 = (tid & 1) * 32;
            const size_t gro = ((size_t)bh * TT + kb * 64 + row) * DHc + c0;
            const uint4* gk = (const uint4*)(Kb + gro);
            *(uint4*)&Ks[row][c0]      = gk[0];
            *(uint4*)&Ks[row][c0 + 8]  = gk[1];
            *(uint4*)&Ks[row][c0 + 16] = gk[2];
            *(uint4*)&Ks[row][c0 + 24] = gk[3];
            union { uint4 q[4]; unsigned short s[32]; } vv;
            const uint4* gv = (const uint4*)(Vb + gro);
            vv.q[0] = gv[0]; vv.q[1] = gv[1]; vv.q[2] = gv[2]; vv.q[3] = gv[3];
#pragma unroll
            for (int j = 0; j < 32; ++j) Vt[c0 + j][row] = vv.s[j];
        }
        __syncthreads();

        // S = (Q * 1/sqrt(dh)) @ K^T : 4 N-tiles x 2 K-steps = 8 WMMAs
        v8f St[4];
#pragma unroll
        for (int nt = 0; nt < 4; ++nt) {
            St[nt] = vzero8();
#pragma unroll
            for (int f = 0; f < 2; ++f) {
                FragBF kf;
                const int br = (nt << 4) + col;      // key within block
                const int kc = f * 32 + (hf << 4);   // dh chunk
                kf.q[0] = *(const uint4*)&Ks[br][kc];
                kf.q[1] = *(const uint4*)&Ks[br][kc + 8];
                St[nt] = __builtin_amdgcn_wmma_f32_16x16x32_bf16(
                    false, qf[f].v, false, kf.v, (short)0, St[nt], false, false);
            }
        }

        // Causal + padding mask (branch-free selects; EXEC untouched)
#pragma unroll
        for (int nt = 0; nt < 4; ++nt) {
            const int kj = kb * 64 + nt * 16 + col;
            const float pm = amask[b * TT + kj] ? 0.f : -1e30f;
#pragma unroll
            for (int r = 0; r < 8; ++r) {
                const int qi = q0 + (w << 4) + r + hf * 8;
                St[nt][r] = (kj <= qi) ? (St[nt][r] + pm) : -1e30f;
            }
        }

        // Online softmax: row-wise max/sum via 16-lane shfl_xor reductions
        float mnew[8], alpha[8];
#pragma unroll
        for (int r = 0; r < 8; ++r) {
            float vmax = fmaxf(fmaxf(St[0][r], St[1][r]),
                               fmaxf(St[2][r], St[3][r]));
#pragma unroll
            for (int off = 1; off < 16; off <<= 1)
                vmax = fmaxf(vmax, __shfl_xor(vmax, off, 32));
            const float mn = fmaxf(mrow[r], vmax);
            alpha[r] = __expf(mrow[r] - mn);
            mnew[r] = mn;
        }
#pragma unroll
        for (int nt = 0; nt < 4; ++nt)
#pragma unroll
            for (int r = 0; r < 8; ++r)
                St[nt][r] = __expf(St[nt][r] - mnew[r]);
#pragma unroll
        for (int r = 0; r < 8; ++r) {
            float s = (St[0][r] + St[1][r]) + (St[2][r] + St[3][r]);
#pragma unroll
            for (int off = 1; off < 16; off <<= 1)
                s += __shfl_xor(s, off, 32);
            lrow[r] = lrow[r] * alpha[r] + s;
            mrow[r] = mnew[r];
        }

        // P: C-layout -> A-layout via wave-private LDS round-trip
#pragma unroll
        for (int nt = 0; nt < 4; ++nt)
#pragma unroll
            for (int r = 0; r < 8; ++r)
                Ps[w][r + hf * 8][nt * 16 + col] = f2bf(St[nt][r]);
        asm volatile("s_wait_dscnt 0x0" ::: "memory");  // wave-local LDS RAW

        // Rescale O, then O += P @ V : 8 WMMAs
#pragma unroll
        for (int nt = 0; nt < 4; ++nt)
#pragma unroll
            for (int r = 0; r < 8; ++r) Ot[nt][r] *= alpha[r];
#pragma unroll
        for (int nt = 0; nt < 4; ++nt) {
#pragma unroll
            for (int f = 0; f < 2; ++f) {
                FragBF pf, vf;
                const int pc = f * 32 + (hf << 4);
                pf.q[0] = *(const uint4*)&Ps[w][col][pc];
                pf.q[1] = *(const uint4*)&Ps[w][col][pc + 8];
                const int vr = (nt << 4) + col;       // dh index
                const int vc = f * 32 + (hf << 4);    // key chunk
                vf.q[0] = *(const uint4*)&Vt[vr][vc];
                vf.q[1] = *(const uint4*)&Vt[vr][vc + 8];
                Ot[nt] = __builtin_amdgcn_wmma_f32_16x16x32_bf16(
                    false, pf.v, false, vf.v, (short)0, Ot[nt], false, false);
            }
        }
    }

    // Finalize: O /= l, store bf16 to y_att in (B,T,D) layout for out-proj
    float inv[8];
#pragma unroll
    for (int r = 0; r < 8; ++r) inv[r] = lrow[r] > 0.f ? 1.f / lrow[r] : 0.f;
#pragma unroll
    for (int nt = 0; nt < 4; ++nt) {
#pragma unroll
        for (int r = 0; r < 8; ++r) {
            const int t = q0 + (w << 4) + r + hf * 8;
            Yb[((size_t)b * TT + t) * DD + h * DHc + nt * 16 + col] =
                f2bf(Ot[nt][r] * inv[r]);
        }
    }
}

// ---------------------------------------------------------------------------
// Launcher
// ---------------------------------------------------------------------------
extern "C" void kernel_launch(void* const* d_in, const int* in_sizes, int n_in,
                              void* d_out, int out_size, void* d_ws,
                              size_t ws_size, hipStream_t stream) {
    const float* x = (const float*)d_in[0];
    const unsigned char* amask = (const unsigned char*)d_in[1];
    const float* w_qkv = (const float*)d_in[2];
    const float* b_qkv = (const float*)d_in[3];
    const float* w_proj = (const float*)d_in[4];
    const float* b_proj = (const float*)d_in[5];
    float* out = (float*)d_out;

    char* ws = (char*)d_ws;
    unsigned short* x_bf    = (unsigned short*)(ws);                      // 16 MB
    unsigned short* wqkv_bf = (unsigned short*)(ws + (((size_t)16) << 20)); //  6 MB
    unsigned short* wprj_bf = (unsigned short*)(ws + (((size_t)22) << 20)); //  2 MB
    unsigned short* Qb      = (unsigned short*)(ws + (((size_t)24) << 20)); // 16 MB
    unsigned short* Kb      = (unsigned short*)(ws + (((size_t)40) << 20)); // 16 MB
    unsigned short* Vb      = (unsigned short*)(ws + (((size_t)56) << 20)); // 16 MB
    unsigned short* Yb      = (unsigned short*)(ws + (((size_t)72) << 20)); // 16 MB

    // 0) fp32 -> bf16
    cvt_f32_to_bf16<<<4096, 256, 0, stream>>>(x, x_bf, (MM * DD) / 4);
    cvt_f32_to_bf16<<<1536, 256, 0, stream>>>(w_qkv, wqkv_bf, (NQKV * DD) / 4);
    cvt_f32_to_bf16<<<512, 256, 0, stream>>>(w_proj, wprj_bf, (DD * DD) / 4);

    // 1) QKV projection + bias + scale + head scatter
    gemm_bf16_wmma<0><<<dim3(MM / 64, NQKV / 64), 128, 0, stream>>>(
        x_bf, wqkv_bf, b_qkv, NQKV, DD, nullptr, Qb, Kb, Vb);

    // 2) causal flash attention
    flash_attn_wmma<<<dim3(TT / 64, BB * HH), 128, 0, stream>>>(Qb, Kb, Vb,
                                                                amask, Yb);

    // 3) output projection + bias -> fp32 d_out
    gemm_bf16_wmma<1><<<dim3(MM / 64, DD / 64), 128, 0, stream>>>(
        Yb, wprj_bf, b_proj, DD, DD, out, nullptr, nullptr, nullptr);
}
